// SelfAttention_65481071407759
// MI455X (gfx1250) — matrix operand order repaired and checked by reading, not compiled, not run on hardware
//
#include <hip/hip_runtime.h>
#include <math.h>

// ---------------------------------------------------------------------------
// Self-attention for MI455X (gfx1250), bf16 WMMA pipeline with async-LDS
// staged B operands.
//   q = x@R, k = x@E, scores = softmax(q k^T / sqrt(D)), out = scores @ x
// B=4, S=4096, D=1024, fp32 in/out; internal math bf16 operands, f32 accum.
// ---------------------------------------------------------------------------

#define NBATCH 4
#define SEQ    4096
#define EMBED  1024

typedef __attribute__((ext_vector_type(16))) __bf16 v16bf;
typedef __attribute__((ext_vector_type(8)))  float  v8f;

union FragBF { v16bf v; uint4 q[2]; };

// round-to-nearest-even f32 -> bf16 (stored as ushort; storage-only)
__device__ __forceinline__ unsigned short f2bf(float x) {
  union { float f; unsigned int u; } v; v.f = x;
  unsigned int r = v.u + 0x7FFFu + ((v.u >> 16) & 1u);
  return (unsigned short)(r >> 16);
}

// Async copy 16B global -> LDS, tracked by ASYNCcnt (CDNA5 VGLOBAL async op).
__device__ __forceinline__ void async_b128(unsigned lds_addr, const void* g) {
  asm volatile("global_load_async_to_lds_b128 %0, %1, off"
               :: "v"(lds_addr), "v"((unsigned long long)g)
               : "memory");
}
__device__ __forceinline__ void wait_async0() {
  asm volatile("s_wait_asynccnt 0x0" ::: "memory");
}

// A fragment, 16x32 bf16 (M x K), row-major source with leading dim ld.
// ISA layout: lane L<16: row M=L, K=0..7 (v0-3) + K=16..23 (v4-7);
//             lane L>=16: row M=L-16, K=8..15 + K=24..31.
__device__ __forceinline__ v16bf load_frag_a(const unsigned short* src, int ld) {
  const int lane = threadIdx.x & 31;
  const int r = lane & 15, h = lane >> 4;
  const unsigned short* p = src + (size_t)r * ld + h * 8;
  FragBF f;
  f.q[0] = *(const uint4*)(p);        // 8 bf16
  f.q[1] = *(const uint4*)(p + 16);   // 8 bf16, +16 elements
  return f.v;
}

// ---------------------------------------------------------------------------
// Cooperative bf16 WMMA GEMM: C[M,N] = scale * (A[M,K] * B[K,N]); B supplied
// as B^T rows (BT[N,K] row-major). Block = 4 waves = 64x64 C tile; each wave
// owns a 16x64 slice. Shared 64x32 B tile double-buffered in LDS, filled by
// global_load_async_to_lds_b128 and consumed with ds_load_b128.
// store_bf16: 1 -> write ushort bf16, 0 -> write f32.
// ---------------------------------------------------------------------------
#define BPITCH 80                       // 64B row + 16B pad -> conflict-free
#define BBUF   (64 * BPITCH)            // one staged 64x32 bf16 tile (5120 B)

__global__ void __launch_bounds__(128)
gemm_bf16_wmma(const unsigned short* __restrict__ A,
               const unsigned short* __restrict__ BT,
               void* __restrict__ Cout,
               int M, int N, int K, int lda, int ldbt, int ldc,
               float scale, int store_bf16) {
  __shared__ unsigned char smem[2 * BBUF];

  const int tid  = (int)threadIdx.x;
  const int w    = tid >> 5;
  const int lane = tid & 31;
  const int ntiles = N >> 6;                        // 64-wide N tiles
  const int n0 = ((int)blockIdx.x % ntiles) << 6;
  const int m0 = ((((int)blockIdx.x / ntiles) << 2) + w) << 4;

  const unsigned smem_base = (unsigned)(uintptr_t)smem;   // LDS byte offset

  // Cooperative async stage of BT[n0..n0+63][kk..kk+31] into buffer `buf`.
  // 128 threads x 32B: thread t covers row t/2, half t&1.
  auto stage = [&](int buf, int kk) {
    const int row = tid >> 1, half = tid & 1;
    const unsigned laddr =
        smem_base + (unsigned)(buf * BBUF + row * BPITCH + half * 32);
    const unsigned short* g = BT + (size_t)(n0 + row) * ldbt + kk + half * 16;
    async_b128(laddr, g);
    async_b128(laddr + 16, g + 8);
  };

  v8f acc[4] = {};
  const unsigned short* arow = A + (size_t)m0 * lda;
  const int bn = lane & 15, bh = lane >> 4;         // B-fragment coords

  stage(0, 0);
  wait_async0();
  __syncthreads();

  int buf = 0;
  for (int kk = 0; kk < K; kk += 32) {
    if (kk + 32 < K) stage(buf ^ 1, kk + 32);       // prefetch next tile

    v16bf a = load_frag_a(arow + kk, lda);
    if (kk + 32 < K) __builtin_prefetch(arow + kk + 32, 0, 1);

    const unsigned char* bbase = smem + buf * BBUF;
#pragma unroll
    for (int j = 0; j < 4; ++j) {
      // B fragment: lane holds column n0+16j+bn, K=8bh..8bh+7 and K=16+8bh..16+8bh+7.
      const unsigned char* p = bbase + (j * 16 + bn) * BPITCH + bh * 16;
      FragBF f;
      f.q[0] = *(const uint4*)(p);
      f.q[1] = *(const uint4*)(p + 32);
      acc[j] = __builtin_amdgcn_wmma_f32_16x16x32_bf16(
          false, a, false, f.v, (short)0, acc[j], false, false);
    }

    wait_async0();      // next tile landed in LDS
    __syncthreads();    // all waves done reading current tile
    buf ^= 1;
  }

  // C/D layout: VGPR r holds (row m0+r, col n0+lane) for lanes 0-15 and
  // (row m0+8+r, col n0+lane-16) for lanes 16-31.
  const int nn = lane & 15;
  const int rb = (lane >> 4) * 8;
#pragma unroll
  for (int j = 0; j < 4; ++j) {
#pragma unroll
    for (int r = 0; r < 8; ++r) {
      float v = acc[j][r] * scale;
      size_t idx = (size_t)(m0 + rb + r) * (size_t)ldc + (size_t)(n0 + 16 * j + nn);
      if (store_bf16) ((unsigned short*)Cout)[idx] = f2bf(v);
      else            ((float*)Cout)[idx] = v;
    }
  }
}

// ---------------------------------------------------------------------------
// Elementwise / transpose conversion kernels
// ---------------------------------------------------------------------------
__global__ void convert_f32_bf16(const float* __restrict__ in,
                                 unsigned short* __restrict__ out, size_t n) {
  for (size_t i = (size_t)blockIdx.x * blockDim.x + threadIdx.x; i < n;
       i += (size_t)gridDim.x * blockDim.x)
    out[i] = f2bf(in[i]);
}

// out[e*D + d] = bf16(in[d*D + e])   (weight transpose)
__global__ void transpose_w_bf16(const float* __restrict__ in,
                                 unsigned short* __restrict__ out, int D) {
  size_t n = (size_t)D * D;
  for (size_t i = (size_t)blockIdx.x * blockDim.x + threadIdx.x; i < n;
       i += (size_t)gridDim.x * blockDim.x) {
    size_t e = i / (size_t)D, d = i % (size_t)D;
    out[i] = f2bf(in[d * (size_t)D + e]);
  }
}

// XT[b][d][s] = bf16(X[b][s][d])   (V^T operand for the P@V GEMM)
__global__ void transpose_x_bf16(const float* __restrict__ in,
                                 unsigned short* __restrict__ out) {
  const size_t per = (size_t)EMBED * SEQ;
  const size_t n = (size_t)NBATCH * per;
  for (size_t i = (size_t)blockIdx.x * blockDim.x + threadIdx.x; i < n;
       i += (size_t)gridDim.x * blockDim.x) {
    size_t b = i / per, rem = i % per;
    size_t d = rem / SEQ, s = rem % SEQ;
    out[i] = f2bf(in[(b * SEQ + s) * EMBED + d]);
  }
}

// ---------------------------------------------------------------------------
// Row softmax: f32 scores in, bf16 probabilities out. One block per row.
// ---------------------------------------------------------------------------
__global__ void __launch_bounds__(256)
softmax_rows(const float* __restrict__ S, unsigned short* __restrict__ P,
             int ncols) {
  __shared__ float red[256];
  const int tid = threadIdx.x;
  const size_t row = blockIdx.x;
  const float* srow = S + row * (size_t)ncols;

  float m = -3.402823466e38f;
  for (int c = tid; c < ncols; c += 256) m = fmaxf(m, srow[c]);
  red[tid] = m; __syncthreads();
  for (int st = 128; st > 0; st >>= 1) {
    if (tid < st) red[tid] = fmaxf(red[tid], red[tid + st]);
    __syncthreads();
  }
  const float rowmax = red[0]; __syncthreads();

  float s = 0.f;
  for (int c = tid; c < ncols; c += 256) s += __expf(srow[c] - rowmax);
  red[tid] = s; __syncthreads();
  for (int st = 128; st > 0; st >>= 1) {
    if (tid < st) red[tid] += red[tid + st];
    __syncthreads();
  }
  const float inv = 1.f / red[0];

  unsigned short* prow = P + row * (size_t)ncols;
  for (int c = tid; c < ncols; c += 256)
    prow[c] = f2bf(__expf(srow[c] - rowmax) * inv);
}

// ---------------------------------------------------------------------------
// Launch
// ---------------------------------------------------------------------------
extern "C" void kernel_launch(void* const* d_in, const int* in_sizes, int n_in,
                              void* d_out, int out_size, void* d_ws, size_t ws_size,
                              hipStream_t stream) {
  (void)in_sizes; (void)n_in; (void)out_size; (void)ws_size;
  const float* X = (const float*)d_in[0];   // [B,S,D]
  const float* R = (const float*)d_in[1];   // [D,D]
  const float* E = (const float*)d_in[2];   // [D,D]
  float* Out = (float*)d_out;               // [B,S,D] f32

  char* ws = (char*)d_ws;
  size_t off = 0;
  auto take = [&](size_t bytes) -> char* {
    char* p = ws + off;
    off += (bytes + 255) & ~(size_t)255;
    return p;
  };
  const size_t NX = (size_t)NBATCH * SEQ * EMBED;   // 16.7M elements
  unsigned short* Xb  = (unsigned short*)take(NX * 2);                  // bf16 X
  unsigned short* Qb  = (unsigned short*)take(NX * 2);                  // bf16 Q
  unsigned short* Kb  = (unsigned short*)take(NX * 2);                  // bf16 K
  unsigned short* XTb = (unsigned short*)take(NX * 2);                  // bf16 X^T per batch
  unsigned short* RTb = (unsigned short*)take((size_t)EMBED * EMBED * 2);
  unsigned short* ETb = (unsigned short*)take((size_t)EMBED * EMBED * 2);
  float*          Sf  = (float*)take((size_t)SEQ * SEQ * 4);            // per-batch scores
  unsigned short* Pb  = (unsigned short*)take((size_t)SEQ * SEQ * 2);   // per-batch probs

  // Stage 0: conversions / transposes
  convert_f32_bf16<<<2048, 256, 0, stream>>>(X, Xb, NX);
  transpose_w_bf16<<<1024, 256, 0, stream>>>(R, RTb, EMBED);
  transpose_w_bf16<<<1024, 256, 0, stream>>>(E, ETb, EMBED);
  transpose_x_bf16<<<2048, 256, 0, stream>>>(X, XTb);

  // Stage 1: projections  Q = X@R, K = X@E   (M=16384, N=K=1024)
  {
    int blocks = ((NBATCH * SEQ) / 64) * (EMBED / 64);   // 4096
    gemm_bf16_wmma<<<blocks, 128, 0, stream>>>(
        Xb, RTb, Qb, NBATCH * SEQ, EMBED, EMBED, EMBED, EMBED, EMBED, 1.f, 1);
    gemm_bf16_wmma<<<blocks, 128, 0, stream>>>(
        Xb, ETb, Kb, NBATCH * SEQ, EMBED, EMBED, EMBED, EMBED, EMBED, 1.f, 1);
  }

  // Stage 2: per-batch attention
  const float scale = 1.0f / 32.0f;   // 1/sqrt(1024)
  for (int b = 0; b < NBATCH; ++b) {
    const unsigned short* Qbb = Qb + (size_t)b * SEQ * EMBED;
    const unsigned short* Kbb = Kb + (size_t)b * SEQ * EMBED;
    // scores = scale * Q K^T : B^T rows are rows of K directly.
    int blocksS = (SEQ / 64) * (SEQ / 64);               // 4096
    gemm_bf16_wmma<<<blocksS, 128, 0, stream>>>(
        Qbb, Kbb, Sf, SEQ, SEQ, EMBED, EMBED, EMBED, SEQ, scale, 0);
    // row softmax -> bf16 P
    softmax_rows<<<SEQ, 256, 0, stream>>>(Sf, Pb, SEQ);
    // out = P @ V, V = X[b]; B^T rows from XT[b].
    int blocksO = (SEQ / 64) * (EMBED / 64);             // 1024
    gemm_bf16_wmma<<<blocksO, 128, 0, stream>>>(
        Pb, XTb + (size_t)b * EMBED * SEQ,
        Out + (size_t)b * SEQ * EMBED,
        SEQ, EMBED, SEQ, SEQ, SEQ, EMBED, 1.f, 0);
  }
}
